// GNN_5480378269924
// MI455X (gfx1250) — matrix operand order repaired and checked by reading, not compiled, not run on hardware
//
#include <hip/hip_runtime.h>
#include <hip/hip_bf16.h>
#include <math.h>
#include <stdint.h>

// ---------------------------------------------------------------------------
// GCN (3x GCNConv + BN + exact GELU) + residual-mean-pool + MLP head, MI455X.
// GEMMs: v_wmma_f32_16x16x32_f16; A tile staged via global_load_async_to_lds
// (ASYNCcnt path, no VGPR roundtrip), converted once to f16 in LDS, fragments
// via ds_load_b128. B pre-transposed to column-major f16 (32B/lane loads).
// Edge aggregation: float4 gather + global_atomic_add_f32 scatter (BW-bound).
// ---------------------------------------------------------------------------

typedef __attribute__((ext_vector_type(16))) _Float16 v16h;
typedef __attribute__((ext_vector_type(8)))  _Float16 v8h;
typedef __attribute__((ext_vector_type(8)))  float    v8f;

#define DEVFN __device__ __forceinline__

DEVFN float gelu_exact(float v) {
    return 0.5f * v * (1.0f + erff(v * 0.70710678118654752f));
}

// ---------------- degree / symmetric norm ----------------------------------
__global__ void k_deg_init(float* __restrict__ deg, int n) {
    int i = blockIdx.x * blockDim.x + threadIdx.x;
    if (i < n) deg[i] = 1.0f;   // self-loop contributes 1
}

__global__ void k_deg_accum(const int* __restrict__ col, float* __restrict__ deg, int e) {
    int i = blockIdx.x * blockDim.x + threadIdx.x;
    int stride = gridDim.x * blockDim.x;
    for (; i < e; i += stride) atomicAdd(&deg[col[i]], 1.0f);
}

__global__ void k_deg_rsqrt(float* __restrict__ deg, int n) {  // in-place -> dinv
    int i = blockIdx.x * blockDim.x + threadIdx.x;
    if (i < n) {
        float d = deg[i];
        deg[i] = d > 0.0f ? rsqrtf(d) : 0.0f;
    }
}

// -------- fp32 row-major [K,Nout] -> f16 column-major [Nout,K] -------------
__global__ void k_cvt_f16_t(const float* __restrict__ W, _Float16* __restrict__ Wt,
                            int K, int Nout) {
    int i = blockIdx.x * blockDim.x + threadIdx.x;
    if (i < K * Nout) {
        int k = i / Nout, c = i - k * Nout;
        Wt[c * K + k] = (_Float16)W[i];
    }
}

// ---------------- WMMA GEMM: Y[n,NOUT] = X[n,K] @ W[K,NOUT] ----------------
// blockDim = 32*(NOUT/16); wave w owns the 16-col tile at w*16.
// Wt is column-major f16 so each lane's B fragment is 16 contiguous halves.
template<int K, int NOUT>
__global__ void __launch_bounds__(256)
k_gemm_wmma_f16(const float* __restrict__ X, const _Float16* __restrict__ Wt,
                float* __restrict__ Y, int n) {
    constexpr int NT = 32 * (NOUT / 16);
    __shared__ __align__(16) float    A32[16 * K];
    __shared__ __align__(16) _Float16 A16[16 * K];
    const int row0 = blockIdx.x * 16;
    const int tid  = threadIdx.x;

    const bool fullTile = (row0 + 16 <= n);   // uniform per block
    if (fullTile) {
        // Async DMA the 16 x K fp32 tile global -> LDS (ASYNCcnt path,
        // no VGPR roundtrip): 16 bytes per lane per issue.
        constexpr int TOTB  = 16 * K * 4;          // tile bytes (4/8 KB)
        constexpr int ITERS = TOTB / (NT * 16);    // 1, 2 or 4 (exact)
        unsigned ldsoff = (unsigned)(uintptr_t)(&A32[0]) + (unsigned)tid * 16u;
        unsigned goff   = (unsigned)tid * 16u;
        const float* gbase = X + (size_t)row0 * K;
#pragma unroll
        for (int it = 0; it < ITERS; ++it) {
            asm volatile("global_load_async_to_lds_b128 %0, %1, %2"
                         :: "v"(ldsoff), "v"(goff), "s"(gbase)
                         : "memory");
            ldsoff += (unsigned)NT * 16u;
            goff   += (unsigned)NT * 16u;
        }
        asm volatile("s_wait_asynccnt 0" ::: "memory");
    } else {
        for (int idx = tid; idx < 16 * K; idx += NT) {
            int r = idx / K, k = idx - r * K;
            int gr = row0 + r;
            A32[idx] = (gr < n) ? X[gr * K + k] : 0.0f;
        }
    }
    __syncthreads();

    // Convert fp32 tile -> f16 tile once (shared by all waves).
    for (int idx = tid * 8; idx < 16 * K; idx += NT * 8) {
        float4 f0 = *reinterpret_cast<const float4*>(&A32[idx]);
        float4 f1 = *reinterpret_cast<const float4*>(&A32[idx + 4]);
        v8h h;
        h[0] = (_Float16)f0.x; h[1] = (_Float16)f0.y;
        h[2] = (_Float16)f0.z; h[3] = (_Float16)f0.w;
        h[4] = (_Float16)f1.x; h[5] = (_Float16)f1.y;
        h[6] = (_Float16)f1.z; h[7] = (_Float16)f1.w;
        *reinterpret_cast<v8h*>(&A16[idx]) = h;
    }
    __syncthreads();

    const int wave = tid >> 5;
    const int lane = tid & 31;
    const int half = lane >> 4;      // 0: lanes 0-15, 1: lanes 16-31
    const int mn   = lane & 15;      // A row (M) / B,C column (N)
    const int c0   = wave * 16;

    const _Float16* __restrict__ arow = &A16[mn * K];
    const _Float16* __restrict__ bcol = &Wt[(c0 + mn) * K];

    v8f acc = {};
#pragma unroll
    for (int kk = 0; kk < K; kk += 32) {
        // A 16x32 f16, k order this code uses (contiguous, same as B): elems 0-7 -> K = kk+half*16 .. +7,
        //                     elems 8-15 -> K = kk+half*16+8 .. +7
        v8h alo = *reinterpret_cast<const v8h*>(arow + kk + half * 16);
        v8h ahi = *reinterpret_cast<const v8h*>(arow + kk + half * 16 + 8);
        v16h a = __builtin_shufflevector(alo, ahi,
                                         0, 1, 2, 3, 4, 5, 6, 7,
                                         8, 9, 10, 11, 12, 13, 14, 15);
        // B 32x16 f16, same contiguous k order as A: lanes 0-15 hold K=kk..kk+15, lanes 16-31 next 16.
        v16h b = *reinterpret_cast<const v16h*>(bcol + kk + half * 16);
        acc = __builtin_amdgcn_wmma_f32_16x16x32_f16(
            /*neg_a=*/false, a, /*neg_b=*/false, b,
            /*c_mod=*/(short)0, acc, /*reuse_a=*/false, /*reuse_b=*/false);
    }

    // C/D 16x16 f32 layout: VGPR g -> row g + half*8, col = mn.
    if (fullTile) {                  // uniform branch: no exec churn
#pragma unroll
        for (int g = 0; g < 8; ++g)
            Y[(row0 + g + half * 8) * NOUT + c0 + mn] = acc[g];
    } else {
#pragma unroll
        for (int g = 0; g < 8; ++g) {
            int r = row0 + g + half * 8;
            if (r < n) Y[r * NOUT + c0 + mn] = acc[g];
        }
    }
}

// ---------------- aggregation: init with bias + self-loop ------------------
template<int D>
__global__ void k_agg_init(const float* __restrict__ H, const float* __restrict__ dinv,
                           const float* __restrict__ bias, float* __restrict__ A, int n) {
    constexpr int D4 = D >> 2;
    int total4 = n * D4;
    int i = blockIdx.x * blockDim.x + threadIdx.x;
    int stride = gridDim.x * blockDim.x;
    for (; i < total4; i += stride) {
        int node = i / D4;                       // power of two -> shift
        int f4   = (i & (D4 - 1)) << 2;
        float w  = dinv[node];
        float w2 = w * w;
        float4 h = *reinterpret_cast<const float4*>(H + node * D + f4);
        float4 o;
        o.x = bias[f4 + 0] + w2 * h.x;
        o.y = bias[f4 + 1] + w2 * h.y;
        o.z = bias[f4 + 2] + w2 * h.z;
        o.w = bias[f4 + 3] + w2 * h.w;
        *reinterpret_cast<float4*>(A + node * D + f4) = o;
    }
}

// ---------------- aggregation: edge gather + atomic scatter ----------------
template<int D>
__global__ void k_edge_scatter(const int* __restrict__ rows, const int* __restrict__ cols,
                               const float* __restrict__ dinv, const float* __restrict__ H,
                               float* __restrict__ A, int e) {
    constexpr int CH = D >> 2;                   // float4 chunks per edge (16/32)
    constexpr int SH = (CH == 32) ? 5 : 4;
    long long idx    = (long long)blockIdx.x * blockDim.x + threadIdx.x;
    long long stride = (long long)gridDim.x * blockDim.x;
    long long total  = (long long)e << SH;
    for (; idx < total; idx += stride) {
        int eid = (int)(idx >> SH);
        int f4  = ((int)idx & (CH - 1)) << 2;
        int r = rows[eid];
        int c = cols[eid];
        int pf = eid + 256; if (pf >= e) pf = e - 1;
        __builtin_prefetch(&rows[pf], 0, 1);     // global_prefetch_b8
        float w = dinv[r] * dinv[c];
        float4 h = *reinterpret_cast<const float4*>(H + r * D + f4);
        float* ap = A + c * D + f4;
        atomicAdd(ap + 0, w * h.x);
        atomicAdd(ap + 1, w * h.y);
        atomicAdd(ap + 2, w * h.z);
        atomicAdd(ap + 3, w * h.w);
    }
}

// ---------------- batch-norm statistics ------------------------------------
__global__ void k_zero(float* __restrict__ p, int count) {
    int i = blockIdx.x * blockDim.x + threadIdx.x;
    if (i < count) p[i] = 0.0f;
}

// stats[0:D] += sum, stats[D:2D] += sumsq. Requires (gridDim*blockDim) % D == 0.
template<int D>
__global__ void k_bn_partial(const float* __restrict__ H, float* __restrict__ stats, int n) {
    int t = blockIdx.x * blockDim.x + threadIdx.x;
    int f = t & (D - 1);
    int r = t / D;
    int rstep = (gridDim.x * blockDim.x) / D;
    float s = 0.0f, q = 0.0f;
    for (; r < n; r += rstep) {
        float v = H[r * D + f];
        s += v;
        q += v * v;
    }
    atomicAdd(&stats[f], s);
    atomicAdd(&stats[D + f], q);
}

// Turn {sum, sumsq} into {scale, shift}: out = h*scale + shift
__global__ void k_bn_finish(float* __restrict__ stats, const float* __restrict__ g,
                            const float* __restrict__ be, int n, int d) {
    int f = threadIdx.x;
    if (f < d) {
        float invn = 1.0f / (float)n;
        float mu   = stats[f] * invn;
        float var  = stats[d + f] * invn - mu * mu;
        float rstd = rsqrtf(var + 1e-5f);
        float scale = g[f] * rstd;
        stats[f]     = scale;
        stats[d + f] = be[f] - mu * scale;
    }
}

template<int D>
__global__ void k_bn_apply_gelu(float* __restrict__ H, const float* __restrict__ stats, int n) {
    constexpr int D4 = D >> 2;
    int total4 = n * D4;
    int i = blockIdx.x * blockDim.x + threadIdx.x;
    int stride = gridDim.x * blockDim.x;
    for (; i < total4; i += stride) {
        int node = i / D4;
        int f4   = (i & (D4 - 1)) << 2;
        float4 h = *reinterpret_cast<const float4*>(H + node * D + f4);
        float4 o;
        o.x = gelu_exact(h.x * stats[f4 + 0] + stats[D + f4 + 0]);
        o.y = gelu_exact(h.y * stats[f4 + 1] + stats[D + f4 + 1]);
        o.z = gelu_exact(h.z * stats[f4 + 2] + stats[D + f4 + 2]);
        o.w = gelu_exact(h.w * stats[f4 + 3] + stats[D + f4 + 3]);
        *reinterpret_cast<float4*>(H + node * D + f4) = o;
    }
}

// ---------------- residual + mean pool (sum only; /n in head) --------------
template<int D>
__global__ void k_pool_partial(const float* __restrict__ X, const float* __restrict__ H,
                               float* __restrict__ pool, int n) {
    int t = blockIdx.x * blockDim.x + threadIdx.x;
    int f = t & (D - 1);
    int r = t / D;
    int rstep = (gridDim.x * blockDim.x) / D;
    float s = 0.0f;
    for (; r < n; r += rstep) s += X[r * D + f] + H[r * D + f];
    atomicAdd(&pool[f], s);
}

// ---------------- MLP head: 64 -> 128 -> 64 -> 32 -> 1 ---------------------
__global__ void k_mlp_head(const float* __restrict__ pool, int n,
                           const float* __restrict__ w1, const float* __restrict__ b1,
                           const float* __restrict__ w2, const float* __restrict__ b2,
                           const float* __restrict__ w3, const float* __restrict__ b3,
                           const float* __restrict__ w4, const float* __restrict__ b4,
                           float* __restrict__ out) {
    __shared__ float v0[64], v1[128], v2[64], v3[32];
    int t = threadIdx.x;
    float invn = 1.0f / (float)n;
    if (t < 64) v0[t] = pool[t] * invn;
    __syncthreads();
    if (t < 128) {
        float s = b1[t];
        for (int k = 0; k < 64; ++k) s += v0[k] * w1[k * 128 + t];
        v1[t] = gelu_exact(s);
    }
    __syncthreads();
    if (t < 64) {
        float s = b2[t];
        for (int k = 0; k < 128; ++k) s += v1[k] * w2[k * 64 + t];
        v2[t] = gelu_exact(s);
    }
    __syncthreads();
    if (t < 32) {
        float s = b3[t];
        for (int k = 0; k < 64; ++k) s += v2[k] * w3[k * 32 + t];
        v3[t] = gelu_exact(s);
    }
    __syncthreads();
    if (t == 0) {
        float s = b4[0];
        for (int k = 0; k < 32; ++k) s += v3[k] * w4[k];
        out[0] = s;
    }
}

// ---------------------------------------------------------------------------
static inline size_t align256(size_t x) { return (x + 255) & ~(size_t)255; }

extern "C" void kernel_launch(void* const* d_in, const int* in_sizes, int n_in,
                              void* d_out, int out_size, void* d_ws, size_t ws_size,
                              hipStream_t stream) {
    const int IN = 64, HID = 128, OUT = 64;
    const int n = in_sizes[0] / IN;
    const int e = in_sizes[1] / 2;

    const float* x    = (const float*)d_in[0];
    const int*   eidx = (const int*)d_in[1];
    const int*   rows = eidx;        // edge_index[0] = gather source
    const int*   cols = eidx + e;    // edge_index[1] = scatter target
    const float* W1 = (const float*)d_in[2],  *b1 = (const float*)d_in[3];
    const float* g1 = (const float*)d_in[4],  *be1 = (const float*)d_in[5];
    const float* W2 = (const float*)d_in[6],  *b2 = (const float*)d_in[7];
    const float* g2 = (const float*)d_in[8],  *be2 = (const float*)d_in[9];
    const float* W3 = (const float*)d_in[10], *b3 = (const float*)d_in[11];
    const float* g3 = (const float*)d_in[12], *be3 = (const float*)d_in[13];
    const float* f1w = (const float*)d_in[14], *f1b = (const float*)d_in[15];
    const float* f2w = (const float*)d_in[16], *f2b = (const float*)d_in[17];
    const float* f3w = (const float*)d_in[18], *f3b = (const float*)d_in[19];
    const float* f4w = (const float*)d_in[20], *f4b = (const float*)d_in[21];
    float* out = (float*)d_out;

    // ---- workspace layout (everything accumulated is re-initialized) ----
    char* ws = (char*)d_ws;
    size_t off = 0;
    const size_t szH = align256((size_t)n * HID * sizeof(float));
    float* hA    = (float*)(ws + off); off += szH;          // GEMM output
    float* hB    = (float*)(ws + off); off += szH;          // aggregation / layer out
    float* dinv  = (float*)(ws + off); off += align256((size_t)n * sizeof(float));
    float* stats = (float*)(ws + off); off += align256(2 * HID * sizeof(float));
    float* pool  = (float*)(ws + off); off += align256(OUT * sizeof(float));
    _Float16* Wt1 = (_Float16*)(ws + off); off += align256((size_t)IN * HID * sizeof(_Float16));
    _Float16* Wt2 = (_Float16*)(ws + off); off += align256((size_t)HID * HID * sizeof(_Float16));
    _Float16* Wt3 = (_Float16*)(ws + off); off += align256((size_t)HID * OUT * sizeof(_Float16));
    (void)ws_size; (void)n_in; (void)out_size;

    const int B = 256;
    const int rowTiles = (n + 15) / 16;

    // ---- degrees -> dinv ----
    k_deg_init<<<(n + B - 1) / B, B, 0, stream>>>(dinv, n);
    k_deg_accum<<<2048, B, 0, stream>>>(cols, dinv, e);
    k_deg_rsqrt<<<(n + B - 1) / B, B, 0, stream>>>(dinv, n);

    // ---- weights: fp32 row-major -> f16 column-major ----
    k_cvt_f16_t<<<(IN * HID + B - 1) / B, B, 0, stream>>>(W1, Wt1, IN, HID);
    k_cvt_f16_t<<<(HID * HID + B - 1) / B, B, 0, stream>>>(W2, Wt2, HID, HID);
    k_cvt_f16_t<<<(HID * OUT + B - 1) / B, B, 0, stream>>>(W3, Wt3, HID, OUT);

    // ---- layer 1: 64 -> 128 ----
    k_gemm_wmma_f16<64, 128><<<rowTiles, 256, 0, stream>>>(x, Wt1, hA, n);
    k_agg_init<128><<<4096, B, 0, stream>>>(hA, dinv, b1, hB, n);
    k_edge_scatter<128><<<8192, B, 0, stream>>>(rows, cols, dinv, hA, hB, e);
    k_zero<<<1, 256, 0, stream>>>(stats, 256);
    k_bn_partial<128><<<2048, B, 0, stream>>>(hB, stats, n);
    k_bn_finish<<<1, 128, 0, stream>>>(stats, g1, be1, n, 128);
    k_bn_apply_gelu<128><<<4096, B, 0, stream>>>(hB, stats, n);

    // ---- layer 2: 128 -> 128 ----
    k_gemm_wmma_f16<128, 128><<<rowTiles, 256, 0, stream>>>(hB, Wt2, hA, n);
    k_agg_init<128><<<4096, B, 0, stream>>>(hA, dinv, b2, hB, n);
    k_edge_scatter<128><<<8192, B, 0, stream>>>(rows, cols, dinv, hA, hB, e);
    k_zero<<<1, 256, 0, stream>>>(stats, 256);
    k_bn_partial<128><<<2048, B, 0, stream>>>(hB, stats, n);
    k_bn_finish<<<1, 128, 0, stream>>>(stats, g2, be2, n, 128);
    k_bn_apply_gelu<128><<<4096, B, 0, stream>>>(hB, stats, n);

    // ---- layer 3: 128 -> 64 ----
    k_gemm_wmma_f16<128, 64><<<rowTiles, 128, 0, stream>>>(hB, Wt3, hA, n);
    k_agg_init<64><<<4096, B, 0, stream>>>(hA, dinv, b3, hB, n);
    k_edge_scatter<64><<<8192, B, 0, stream>>>(rows, cols, dinv, hA, hB, e);
    k_zero<<<1, 256, 0, stream>>>(stats, 128);
    k_bn_partial<64><<<2048, B, 0, stream>>>(hB, stats, n);
    k_bn_finish<<<1, 64, 0, stream>>>(stats, g3, be3, n, 64);
    k_bn_apply_gelu<64><<<4096, B, 0, stream>>>(hB, stats, n);

    // ---- residual + mean pool + head ----
    k_zero<<<1, OUT, 0, stream>>>(pool, OUT);
    k_pool_partial<64><<<2048, B, 0, stream>>>(x, hB, pool, n);
    k_mlp_head<<<1, 128, 0, stream>>>(pool, n, f1w, f1b, f2w, f2b, f3w, f3b, f4w, f4b, out);
}
